// AmrArl_17540646437037
// MI455X (gfx1250) — matrix operand order repaired and checked by reading, not compiled, not run on hardware
//
#include <hip/hip_runtime.h>
#include <hip/hip_bf16.h>

typedef __attribute__((ext_vector_type(16))) _Float16 v16h;
typedef __attribute__((ext_vector_type(8)))  _Float16 v8h;
typedef __attribute__((ext_vector_type(8)))  float    v8f;

#define NUM_ASPECTS 5
#define CTX 3
#define H1 10
#define WORD_DIM 300
#define REVIEW_LEN 500
#define BATCH 512
#define NCOLS (NUM_ASPECTS * H1)   // 50 valid output columns
#define NPAD 64                    // padded to 4 N-tiles of 16
#define KPAD 320                   // 300 padded to 10 K-steps of 32
#define WT_STRIDE 328              // LDS row stride (f16) -> conflict-free B reads

// ---------------------------------------------------------------------------
// Kernel 1: proj[row][n] = sum_k emb[row][k] * W[k][n]
//   rows = BATCH*REVIEW_LEN = 256000 (multiple of 16), W = asp_proj (a,d,h)
//   f16 WMMA (v_wmma_f32_16x16x32_f16), f32 accumulate.
// ---------------------------------------------------------------------------
__global__ __launch_bounds__(128)
void proj_gemm_wmma(const float* __restrict__ emb,       // (256000, 300)
                    const float* __restrict__ asp_proj,  // (5, 300, 10)
                    float* __restrict__ proj)            // (256000, 50)
{
    __shared__ _Float16 s_wt[NPAD * WT_STRIDE];  // Wt[n][k], f16, zero-padded

    const int tid  = threadIdx.x;
    const int lane = tid & 31;
    const int wave = tid >> 5;

    // Stage transposed weights in LDS (zero-padded to 64 x 320).
    for (int idx = tid; idx < NPAD * KPAD; idx += 128) {
        const int n = idx / KPAD;
        const int k = idx - n * KPAD;
        float v = 0.0f;
        if (n < NCOLS && k < WORD_DIM) {
            const int a = n / H1, h = n - a * H1;
            v = asp_proj[((size_t)a * WORD_DIM + k) * H1 + h];
        }
        s_wt[n * WT_STRIDE + k] = (_Float16)v;
    }
    __syncthreads();

    const int tile = blockIdx.x * 4 + wave;   // 16-row M-tile per wave
    const int m    = lane & 15;               // A: row within tile / B: column
    const int half = lane >> 4;
    const size_t row = (size_t)tile * 16 + m;
    const float* __restrict__ arow = emb + row * WORD_DIM;

    v8f acc[4];
    #pragma unroll
    for (int nt = 0; nt < 4; ++nt) acc[nt] = (v8f){0.f,0.f,0.f,0.f,0.f,0.f,0.f,0.f};

    for (int kb = 0; kb < KPAD; kb += 32) {
        // ---- A fragment (16x32 f16, ISA layout): two contiguous 8-K groups
        const int k0 = kb + half * 8;        // elements 0..7  -> K = k0 .. k0+7
        const int k1 = kb + 16 + half * 8;   // elements 8..15 -> K = k1 .. k1+7
        v16h a;
        if (kb + 31 < WORD_DIM) {
            const float4 f0 = *(const float4*)(arow + k0);
            const float4 f1 = *(const float4*)(arow + k0 + 4);
            const float4 f2 = *(const float4*)(arow + k1);
            const float4 f3 = *(const float4*)(arow + k1 + 4);
            a[0]=(_Float16)f0.x; a[1]=(_Float16)f0.y; a[2]=(_Float16)f0.z; a[3]=(_Float16)f0.w;
            a[4]=(_Float16)f1.x; a[5]=(_Float16)f1.y; a[6]=(_Float16)f1.z; a[7]=(_Float16)f1.w;
            a[8]=(_Float16)f2.x; a[9]=(_Float16)f2.y; a[10]=(_Float16)f2.z; a[11]=(_Float16)f2.w;
            a[12]=(_Float16)f3.x; a[13]=(_Float16)f3.y; a[14]=(_Float16)f3.z; a[15]=(_Float16)f3.w;
        } else {                              // final partial K-step (kb == 288)
            #pragma unroll
            for (int e = 0; e < 8; ++e) {
                a[e]     = (_Float16)((k0 + e) < WORD_DIM ? arow[k0 + e] : 0.0f);
                a[e + 8] = (_Float16)((k1 + e) < WORD_DIM ? arow[k1 + e] : 0.0f);
            }
        }

        // ---- B fragments (32x16 f16): lanes 0-15 hold K 0-7,16-23; 16-31 hold K 8-15,24-31
        #pragma unroll
        for (int nt = 0; nt < 4; ++nt) {
            const _Float16* wrow = &s_wt[(nt * 16 + m) * WT_STRIDE + kb + half * 8];
            const v8h b0 = *(const v8h*)(wrow);
            const v8h b1 = *(const v8h*)(wrow + 16);
            v16h b;
            #pragma unroll
            for (int e = 0; e < 8; ++e) { b[e] = b0[e]; b[e + 8] = b1[e]; }
            acc[nt] = __builtin_amdgcn_wmma_f32_16x16x32_f16(
                false, a, false, b, (short)0, acc[nt], false, false);
        }
    }

    // ---- Store C/D (lane = N within tile, VGPR r -> M = r + 8*half)
    #pragma unroll
    for (int nt = 0; nt < 4; ++nt) {
        const int n = nt * 16 + m;
        if (n < NCOLS) {
            #pragma unroll
            for (int r = 0; r < 8; ++r) {
                const int M = r + 8 * half;
                proj[((size_t)tile * 16 + M) * NCOLS + n] = acc[nt][r];
            }
        }
    }
}

// ---------------------------------------------------------------------------
// Kernel 2: windowed context score + softmax over L + weighted pooling.
//   One block per (b, a). Deterministic fixed-order reductions.
// ---------------------------------------------------------------------------
__global__ __launch_bounds__(256)
void attn_pool(const float* __restrict__ proj,       // (256000, 50)
               const float* __restrict__ asp_embed,  // (5, 30)
               float* __restrict__ out_attn,         // (512, 5, 500)
               float* __restrict__ out_rep)          // (512, 5, 10)
{
    __shared__ float s_proj[(REVIEW_LEN + 2) * H1];  // zero-padded window rows
    __shared__ float s_val[REVIEW_LEN];              // score -> exp values
    __shared__ float red[256];
    __shared__ float s_part[8 * H1];                 // per-wave pooling partials

    const int b   = blockIdx.x / NUM_ASPECTS;
    const int a   = blockIdx.x % NUM_ASPECTS;
    const int tid = threadIdx.x;

    float E[CTX * H1];
    #pragma unroll
    for (int i = 0; i < CTX * H1; ++i) E[i] = asp_embed[a * (CTX * H1) + i];

    for (int idx = tid; idx < (REVIEW_LEN + 2) * H1; idx += 256) {
        const int r = idx / H1, h = idx - r * H1;
        float v = 0.0f;
        if (r >= 1 && r <= REVIEW_LEN)
            v = proj[((size_t)b * REVIEW_LEN + (r - 1)) * NCOLS + a * H1 + h];
        s_proj[idx] = v;
    }
    __syncthreads();

    // score[l] = sum_{w,h} s_proj[l+w][h] * E[w*H1+h]
    for (int l = tid; l < REVIEW_LEN; l += 256) {
        float s = 0.0f;
        #pragma unroll
        for (int w = 0; w < CTX; ++w)
            #pragma unroll
            for (int h = 0; h < H1; ++h)
                s += s_proj[(l + w) * H1 + h] * E[w * H1 + h];
        s_val[l] = s;
    }
    __syncthreads();

    // softmax max (tree reduce)
    float lm = -3.402823466e38f;
    for (int l = tid; l < REVIEW_LEN; l += 256) lm = fmaxf(lm, s_val[l]);
    red[tid] = lm; __syncthreads();
    for (int s = 128; s > 0; s >>= 1) {
        if (tid < s) red[tid] = fmaxf(red[tid], red[tid + s]);
        __syncthreads();
    }
    const float mx = red[0];
    __syncthreads();

    // softmax sum
    float ls = 0.0f;
    for (int l = tid; l < REVIEW_LEN; l += 256) {
        const float e = expf(s_val[l] - mx);
        s_val[l] = e;
        ls += e;
    }
    red[tid] = ls; __syncthreads();
    for (int s = 128; s > 0; s >>= 1) {
        if (tid < s) red[tid] += red[tid + s];
        __syncthreads();
    }
    const float inv = 1.0f / red[0];
    __syncthreads();

    // attn out + pooling partials
    float rh[H1];
    #pragma unroll
    for (int h = 0; h < H1; ++h) rh[h] = 0.0f;
    for (int l = tid; l < REVIEW_LEN; l += 256) {
        const float at = s_val[l] * inv;
        out_attn[((size_t)b * NUM_ASPECTS + a) * REVIEW_LEN + l] = at;
        #pragma unroll
        for (int h = 0; h < H1; ++h) rh[h] += at * s_proj[(l + 1) * H1 + h];
    }
    // wave32 shuffle reduce, then fixed-order cross-wave sum
    #pragma unroll
    for (int h = 0; h < H1; ++h) {
        float v = rh[h];
        for (int off = 16; off > 0; off >>= 1) v += __shfl_down(v, off, 32);
        if ((tid & 31) == 0) s_part[(tid >> 5) * H1 + h] = v;
    }
    __syncthreads();
    if (tid < H1) {
        float s = 0.0f;
        #pragma unroll
        for (int w = 0; w < 8; ++w) s += s_part[w * H1 + tid];
        out_rep[((size_t)b * NUM_ASPECTS + a) * H1 + tid] = s;
    }
}

// ---------------------------------------------------------------------------
extern "C" void kernel_launch(void* const* d_in, const int* in_sizes, int n_in,
                              void* d_out, int out_size, void* d_ws, size_t ws_size,
                              hipStream_t stream) {
    const float* review_emb = (const float*)d_in[0];  // (512, 500, 300)
    const float* asp_embed  = (const float*)d_in[1];  // (5, 30)
    const float* asp_proj   = (const float*)d_in[2];  // (5, 300, 10)

    float* out_attn = (float*)d_out;                                     // (512,5,500)
    float* out_rep  = out_attn + (size_t)BATCH * NUM_ASPECTS * REVIEW_LEN; // (512,5,10)
    float* proj     = (float*)d_ws;  // (256000, 50) f32 = 51.2 MB scratch

    const int rows   = BATCH * REVIEW_LEN;      // 256000
    const int mtiles = rows / 16;               // 16000
    proj_gemm_wmma<<<mtiles / 4, 128, 0, stream>>>(review_emb, asp_proj, proj);
    attn_pool<<<BATCH * NUM_ASPECTS, 256, 0, stream>>>(proj, asp_embed, out_attn, out_rep);
}
